// HeteroVGAE_9285719294036
// MI455X (gfx1250) — compile-verified, matrix operand order checked by reading, not run on hardware
//
#include <hip/hip_runtime.h>

#define ND   8192
#define NG   16384
#define EE   524288
#define FD   256
#define H1   64
#define H2   32
#define OUTD 64

typedef __attribute__((ext_vector_type(16))) _Float16 v16h;
typedef __attribute__((ext_vector_type(8)))  float    v8f;

// ISA 7.12.2: 16-bit A-matrix 16x32 fragment K index for (lane, half h).
// Lanes 0-15 hold K {0..7, 16..23}; lanes 16-31 hold K {8..15, 24..31}.
// B (32x16) uses the mirrored layout with lane = column.
// Per lane this is two contiguous 16-byte runs -> compiler emits b128 loads.
__device__ __forceinline__ int frag_k(int lane, int h) {
  int hi = (lane >= 16) ? 8 : 0;
  return (h < 8) ? (h + hi) : (16 + (h - 8) + hi);
}

// ---------------- edge kernels ----------------

__global__ void count_kernel(const int* __restrict__ dst, float* __restrict__ cnt, int E) {
  int e = blockIdx.x * blockDim.x + threadIdx.x;
  if (e < E) atomicAdd(&cnt[dst[e]], 1.0f);
}

template <int F>
__global__ void scatter_kernel(const float* __restrict__ xsrc, const int* __restrict__ src,
                               const int* __restrict__ dst, float* __restrict__ agg, int E) {
  const int CPE = F / 4;  // float4 chunks per edge
  int idx = blockIdx.x * blockDim.x + threadIdx.x;
  if (idx >= E * CPE) return;
  int e = idx / CPE;
  int c = (idx % CPE) * 4;
  int s = src[e], d = dst[e];
  const float4 v = *(const float4*)(xsrc + (size_t)s * F + c);
  float* a = agg + (size_t)d * F + c;
  atomicAdd(a + 0, v.x);
  atomicAdd(a + 1, v.y);
  atomicAdd(a + 2, v.z);
  atomicAdd(a + 3, v.w);
}

// ---------------- SAGE linear:  h = (agg/cnt) @ Wl^T + bl + xdst @ Wr^T ----------------
// One wave per 16-node tile; HOUT/16 WMMA column tiles; K loop fully unrolled.

template <int F, int HOUT>
__global__ __launch_bounds__(32) void sage_lin_kernel(
    const float* __restrict__ agg, const float* __restrict__ cnt,
    const float* __restrict__ xdst,
    const float* __restrict__ Wl, const float* __restrict__ bl,
    const float* __restrict__ Wr, float* __restrict__ hout) {
  constexpr int NT = HOUT / 16;
  const int lane = threadIdx.x;
  const int tile = blockIdx.x;
  const int row  = lane & 15;
  const int node = tile * 16 + row;
  const float inv = 1.0f / fmaxf(cnt[node], 1.0f);

  v8f acc[NT] = {};

  // part 1: mean-aggregated neighbors @ Wl^T   (scale applied on load)
#pragma unroll
  for (int k0 = 0; k0 < F; k0 += 32) {
    v16h a;
#pragma unroll
    for (int h = 0; h < 16; ++h)
      a[h] = (_Float16)(agg[(size_t)node * F + k0 + frag_k(lane, h)] * inv);
#pragma unroll
    for (int t = 0; t < NT; ++t) {
      const int col = t * 16 + row;
      v16h b;
#pragma unroll
      for (int h = 0; h < 16; ++h)
        b[h] = (_Float16)(Wl[col * F + k0 + frag_k(lane, h)]);
      acc[t] = __builtin_amdgcn_wmma_f32_16x16x32_f16(false, a, false, b,
                                                      (short)0, acc[t], false, false);
    }
  }
  // part 2: self features @ Wr^T
#pragma unroll
  for (int k0 = 0; k0 < F; k0 += 32) {
    v16h a;
#pragma unroll
    for (int h = 0; h < 16; ++h)
      a[h] = (_Float16)(xdst[(size_t)node * F + k0 + frag_k(lane, h)]);
#pragma unroll
    for (int t = 0; t < NT; ++t) {
      const int col = t * 16 + row;
      v16h b;
#pragma unroll
      for (int h = 0; h < 16; ++h)
        b[h] = (_Float16)(Wr[col * F + k0 + frag_k(lane, h)]);
      acc[t] = __builtin_amdgcn_wmma_f32_16x16x32_f16(false, a, false, b,
                                                      (short)0, acc[t], false, false);
    }
  }
  // C/D layout: VGPR r holds M = r + 8*(lane>=16), N = lane%16
  const int mhi = (lane >= 16) ? 8 : 0;
#pragma unroll
  for (int t = 0; t < NT; ++t) {
    const int col = t * 16 + row;
    const float bv = bl[col];
#pragma unroll
    for (int r = 0; r < 8; ++r) {
      const int m = r + mhi;
      hout[(size_t)(tile * 16 + m) * HOUT + col] = acc[t][r] + bv;
    }
  }
}

// ---------------- VGAE heads + reparametrize, quantize to f16 ----------------

__global__ void heads_kernel(const float* __restrict__ z,
                             const float* __restrict__ Wmu, const float* __restrict__ bmu,
                             const float* __restrict__ Wlv, const float* __restrict__ blv,
                             const float* __restrict__ eps,
                             _Float16* __restrict__ zh, int Nn) {
  int idx = blockIdx.x * blockDim.x + threadIdx.x;
  if (idx >= Nn * OUTD) return;
  int n = idx / OUTD, o = idx % OUTD;
  const float* zr = z + (size_t)n * H2;
  float mu = bmu[o], ls = blv[o];
#pragma unroll
  for (int k = 0; k < H2; ++k) {
    mu += zr[k] * Wmu[o * H2 + k];
    ls += zr[k] * Wlv[o * H2 + k];
  }
  zh[idx] = (_Float16)(mu + eps[idx] * expf(ls));
}

// ---------------- decoder: out[ND,NG] = zd @ zg^T (K=64) ----------------
// 4 waves/block; each wave computes a 16x64 output strip (4 WMMA col tiles),
// A fragment loaded once per K-chunk and reused across all 4 tiles.
// Stores ordered per row so 4x64B segments are address-consecutive (256B bursts).

__global__ __launch_bounds__(128) void decoder_kernel(
    const _Float16* __restrict__ zd, const _Float16* __restrict__ zg,
    float* __restrict__ out) {
  const int lane = threadIdx.x & 31;
  const int wv   = threadIdx.x >> 5;
  const int rowTile = blockIdx.y * 4 + wv;  // ND/16 row tiles total
  const int colBase = blockIdx.x * 64;      // NG/64 column strips
  const int row  = lane & 15;
  const int mrow = rowTile * 16 + row;      // A row (disease)

  v8f acc[4] = {};
#pragma unroll
  for (int k0 = 0; k0 < OUTD; k0 += 32) {
    v16h a;
#pragma unroll
    for (int h = 0; h < 16; ++h)
      a[h] = zd[(size_t)mrow * OUTD + k0 + frag_k(lane, h)];
#pragma unroll
    for (int t = 0; t < 4; ++t) {
      const int ncol = colBase + t * 16 + row;  // B column = zg row (gene)
      v16h b;
#pragma unroll
      for (int h = 0; h < 16; ++h)
        b[h] = zg[(size_t)ncol * OUTD + k0 + frag_k(lane, h)];
      acc[t] = __builtin_amdgcn_wmma_f32_16x16x32_f16(false, a, false, b,
                                                      (short)0, acc[t], false, false);
    }
  }
  const int mhi = (lane >= 16) ? 8 : 0;
#pragma unroll
  for (int r = 0; r < 8; ++r) {
    const size_t m = (size_t)rowTile * 16 + r + mhi;
    float* o = out + m * NG + colBase + row;
#pragma unroll
    for (int t = 0; t < 4; ++t)
      o[t * 16] = acc[t][r];
  }
}

// ---------------- launch ----------------

extern "C" void kernel_launch(void* const* d_in, const int* in_sizes, int n_in,
                              void* d_out, int out_size, void* d_ws, size_t ws_size,
                              hipStream_t stream) {
  (void)in_sizes; (void)n_in; (void)out_size; (void)ws_size;
  const float* x_d     = (const float*)d_in[0];
  const float* x_g     = (const float*)d_in[1];
  const float* W1_dg_l = (const float*)d_in[2];
  const float* b1_dg_l = (const float*)d_in[3];
  const float* W1_dg_r = (const float*)d_in[4];
  const float* W1_gd_l = (const float*)d_in[5];
  const float* b1_gd_l = (const float*)d_in[6];
  const float* W1_gd_r = (const float*)d_in[7];
  const float* W2_dg_l = (const float*)d_in[8];
  const float* b2_dg_l = (const float*)d_in[9];
  const float* W2_dg_r = (const float*)d_in[10];
  const float* W2_gd_l = (const float*)d_in[11];
  const float* b2_gd_l = (const float*)d_in[12];
  const float* W2_gd_r = (const float*)d_in[13];
  const float* Wmu_d   = (const float*)d_in[14];
  const float* bmu_d   = (const float*)d_in[15];
  const float* Wlv_d   = (const float*)d_in[16];
  const float* blv_d   = (const float*)d_in[17];
  const float* Wmu_g   = (const float*)d_in[18];
  const float* bmu_g   = (const float*)d_in[19];
  const float* Wlv_g   = (const float*)d_in[20];
  const float* blv_g   = (const float*)d_in[21];
  const float* eps_d   = (const float*)d_in[22];
  const float* eps_g   = (const float*)d_in[23];
  const int*   e_dg_s  = (const int*)d_in[24];
  const int*   e_dg_d  = (const int*)d_in[25];
  const int*   e_gd_s  = (const int*)d_in[26];
  const int*   e_gd_d  = (const int*)d_in[27];
  float* out = (float*)d_out;

  // workspace layout (floats); accumulation region first so one memset clears it
  float* base = (float*)d_ws;
  size_t off = 0;
  float* cnt_g  = base + off; off += NG;
  float* cnt_d  = base + off; off += ND;
  float* agg_g1 = base + off; off += (size_t)NG * FD;
  float* agg_d1 = base + off; off += (size_t)ND * FD;
  float* agg_g2 = base + off; off += (size_t)NG * H1;
  float* agg_d2 = base + off; off += (size_t)ND * H1;
  const size_t zero_bytes = off * sizeof(float);
  float* h_g = base + off; off += (size_t)NG * H1;
  float* h_d = base + off; off += (size_t)ND * H1;
  float* z_g = base + off; off += (size_t)NG * H2;
  float* z_d = base + off; off += (size_t)ND * H2;
  _Float16* zd_h = (_Float16*)(base + off); off += (size_t)ND * OUTD / 2;
  _Float16* zg_h = (_Float16*)(base + off); off += (size_t)NG * OUTD / 2;

  hipMemsetAsync(d_ws, 0, zero_bytes, stream);

  // degree counts (shared by both layers: same edge lists)
  count_kernel<<<EE / 256, 256, 0, stream>>>(e_dg_d, cnt_g, EE);
  count_kernel<<<EE / 256, 256, 0, stream>>>(e_gd_d, cnt_d, EE);

  // layer 1 scatter-add (256 features, float4 per thread; L2-resident targets)
  scatter_kernel<FD><<<(EE * (FD / 4)) / 256, 256, 0, stream>>>(x_d, e_dg_s, e_dg_d, agg_g1, EE);
  scatter_kernel<FD><<<(EE * (FD / 4)) / 256, 256, 0, stream>>>(x_g, e_gd_s, e_gd_d, agg_d1, EE);

  // layer 1 SAGE linear (WMMA)
  sage_lin_kernel<FD, H1><<<NG / 16, 32, 0, stream>>>(agg_g1, cnt_g, x_g, W1_dg_l, b1_dg_l, W1_dg_r, h_g);
  sage_lin_kernel<FD, H1><<<ND / 16, 32, 0, stream>>>(agg_d1, cnt_d, x_d, W1_gd_l, b1_gd_l, W1_gd_r, h_d);

  // layer 2 scatter-add (64 features)
  scatter_kernel<H1><<<(EE * (H1 / 4)) / 256, 256, 0, stream>>>(h_d, e_dg_s, e_dg_d, agg_g2, EE);
  scatter_kernel<H1><<<(EE * (H1 / 4)) / 256, 256, 0, stream>>>(h_g, e_gd_s, e_gd_d, agg_d2, EE);

  // layer 2 SAGE linear (WMMA)
  sage_lin_kernel<H1, H2><<<NG / 16, 32, 0, stream>>>(agg_g2, cnt_g, h_g, W2_dg_l, b2_dg_l, W2_dg_r, z_g);
  sage_lin_kernel<H1, H2><<<ND / 16, 32, 0, stream>>>(agg_d2, cnt_d, h_d, W2_gd_l, b2_gd_l, W2_gd_r, z_d);

  // mu/logstd heads + reparametrize, quantize to f16 for decoder
  heads_kernel<<<(ND * OUTD) / 256, 256, 0, stream>>>(z_d, Wmu_d, bmu_d, Wlv_d, blv_d, eps_d, zd_h, ND);
  heads_kernel<<<(NG * OUTD) / 256, 256, 0, stream>>>(z_g, Wmu_g, bmu_g, Wlv_g, blv_g, eps_g, zg_h, NG);

  // inner-product decoder (WMMA, store-bandwidth bound: 536 MB f32)
  decoder_kernel<<<dim3(NG / 64, ND / 64), 128, 0, stream>>>(zd_h, zg_h, out);
}
